// DoubleJpeg_81423989997862
// MI455X (gfx1250) — matrix operand order repaired and checked by reading, not compiled, not run on hardware
//
#include <hip/hip_runtime.h>
#include <hip/hip_bf16.h>

typedef __attribute__((ext_vector_type(16))) _Float16 v16h;
typedef __attribute__((ext_vector_type(8)))  _Float16 v8h;
typedef __attribute__((ext_vector_type(8)))  float    v8f;

// Combine two 8-half LDS chunks (p, p+gap) into one 16-half WMMA fragment.
__device__ __forceinline__ v16h ld_pair8(const _Float16* p, int gap) {
  v8h lo = *(const v8h*)p;
  v8h hi = *(const v8h*)(p + gap);
  return __builtin_shufflevector(lo, hi, 0,1,2,3,4,5,6,7,8,9,10,11,12,13,14,15);
}

// ---------------------------------------------------------------------------
// 1) 8x8 block DCT: x[64,1,256,256] -> xd[64,64ch,32,32] (f32)
// ---------------------------------------------------------------------------
__global__ void dct_kernel(const float* __restrict__ x,
                           const float* __restrict__ basis,  // [64 freq][64 pix]
                           float* __restrict__ xd) {
  __shared__ float tile[4][64];
  const int lt = threadIdx.x & 63;   // lane in tile (freq / pixel)
  const int st = threadIdx.x >> 6;   // subtile 0..3
  const int tIdx = blockIdx.x * 4 + st;        // 0..65535
  const int b  = tIdx >> 10;
  const int r  = tIdx & 1023;                  // tile within image
  const int ty = r >> 5, tx = r & 31;
  const int py = lt >> 3, px = lt & 7;
  tile[st][lt] = x[((b * 256) + ty * 8 + py) * 256 + tx * 8 + px];
  __syncthreads();
  float acc = 0.f;
  const float* bs = basis + lt * 64;
  #pragma unroll 8
  for (int j = 0; j < 64; ++j) acc += bs[j] * tile[st][j];
  xd[((b * 64 + lt) * 1024) + r] = acc;
}

// ---------------------------------------------------------------------------
// 2) 121-bin soft histogram -> per-bin counts: feat[B,120,64] (f32)
// ---------------------------------------------------------------------------
__global__ void hist_kernel(const float* __restrict__ xd,
                            float* __restrict__ feat) {
  __shared__ float vals[1024];
  __shared__ float hist[122];
  const int plane = blockIdx.x;        // b*64 + ch
  const int b = plane >> 6, ch = plane & 63;
  const float* src = xd + (long)plane * 1024;
  for (int i = threadIdx.x; i < 1024; i += 128) vals[i] = src[i];
  __syncthreads();
  const int k = threadIdx.x;
  if (k < 121) {
    const float bin = (float)(k - 60);
    float s = 0.f;
    for (int i = 0; i < 1024; ++i) {
      float t = 1.0e6f * (vals[i] - bin);
      s += 1.0f / (1.0f + __expf(-t));   // saturates cleanly for |t| large
    }
    hist[k] = s * (1.0f / 1024.0f);
  }
  __syncthreads();
  if (k < 120) feat[((b * 120) + k) * 64 + ch] = hist[k] - hist[k + 1];
}

// ---------------------------------------------------------------------------
// 3) conv1a: Cin=1 direct conv, feat f32 -> act f16 NHWC [B,120,64,64]
// ---------------------------------------------------------------------------
__global__ void conv1a_kernel(const float* __restrict__ feat,
                              const float* __restrict__ w,   // [64][25]
                              const float* __restrict__ bias,
                              _Float16* __restrict__ out) {
  const int idx = blockIdx.x * 256 + threadIdx.x;  // ((b*120+y)*64+x)*64+oc
  if (idx >= 64 * 120 * 64 * 64) return;
  const int oc = idx & 63;
  const int p  = idx >> 6;
  const int x  = p & 63;
  const int y  = (p >> 6) % 120;
  const int b  = p / (120 * 64);
  float acc = bias[oc];
  const float* wr = w + oc * 25;
  #pragma unroll
  for (int t = 0; t < 25; ++t) {
    const int iy = y + t / 5 - 2, ix = x + (t % 5) - 2;
    if (iy >= 0 && iy < 120 && ix >= 0 && ix < 64)
      acc += wr[t] * feat[(b * 120 + iy) * 64 + ix];
  }
  out[idx] = (_Float16)acc;
}

// ---------------------------------------------------------------------------
// Weight repack: OIHW f32 -> [O][tap*Cin + ci] f16 (K-order = (ky,kx,ci))
// ---------------------------------------------------------------------------
__global__ void pack_w_kernel(const float* __restrict__ w, _Float16* __restrict__ wq,
                              int O, int I) {
  const int K = 25 * I;
  const long idx = (long)blockIdx.x * 256 + threadIdx.x;
  if (idx >= (long)O * K) return;
  const int o = (int)(idx / K), k = (int)(idx % K);
  const int tap = k / I, ci = k % I;
  wq[idx] = (_Float16)w[(o * I + ci) * 25 + tap];
}

// ---------------------------------------------------------------------------
// 4) Implicit-GEMM conv via v_wmma_f32_16x16x32_f16.
//    Block: 128 thr / 4 waves; 64 outch x 64 positions; double-buffered LDS,
//    one barrier per K-chunk, next chunk's global loads issued before WMMAs.
//    Dims are template constants -> no runtime div/mod, constant trip counts.
// ---------------------------------------------------------------------------
template <int H, int W, int Cin, int Cout>
__global__ void __launch_bounds__(128) conv_wmma_kernel(
    const _Float16* __restrict__ in,   // NHWC [B][H][W][Cin]
    const _Float16* __restrict__ wq,   // [Cout][25*Cin]
    const float* __restrict__ bias,
    _Float16* __restrict__ out) {     // NHWC [B][H][W][Cout]
  constexpr int K   = 25 * Cin;
  constexpr int NCH = K / 32;        // K-chunks
  constexpr int CPT = Cin / 32;      // chunks per 5x5 tap
  __shared__ __align__(16) _Float16 sA[2][64 * 32];
  __shared__ __align__(16) _Float16 sB[2][64 * 32];
  const int tid  = threadIdx.x;
  const int lane = tid & 31, wave = tid >> 5;
  const int m0blk = blockIdx.y * 64;
  const int pbase = blockIdx.x * 64;

  const int rowS = tid >> 1, hh = tid & 1;   // staging: row 0..63, half 0/1
  const int pidx = pbase + rowS;             // this row's output position
  const int xs  = pidx % W;                  // W is a power of two here
  const int ys  = (pidx / W) % H;
  const int bs_ = pidx / (W * H);

  const int n     = lane & 15;
  const int koffA = (lane < 16) ? 0 : 8;
  const int koffB = (lane < 16) ? 0 : 8;
  const int mRowA = wave * 16 + n;

  auto stage = [&](int c, int buf) {
    // A: 64 outch rows x 32 K of the weight matrix
    const _Float16* asrc = wq + (m0blk + rowS) * K + c * 32 + hh * 16;
    *(v8h*)&sA[buf][rowS * 32 + hh * 16]     = *(const v8h*)asrc;
    *(v8h*)&sA[buf][rowS * 32 + hh * 16 + 8] = *(const v8h*)(asrc + 8);
    // B (im2col): 64 positions x 32 channels of one tap
    const int tap = c / CPT;
    const int ci0 = (c % CPT) << 5;
    const int iy = ys + tap / 5 - 2, ix = xs + tap % 5 - 2;
    v8h v0 = {}, v1 = {};
    if (iy >= 0 && iy < H && ix >= 0 && ix < W) {
      const _Float16* bsrc =
          in + ((long)((bs_ * H + iy) * W + ix)) * Cin + ci0 + hh * 16;
      v0 = *(const v8h*)bsrc;
      v1 = *(const v8h*)(bsrc + 8);
    }
    *(v8h*)&sB[buf][rowS * 32 + hh * 16]     = v0;
    *(v8h*)&sB[buf][rowS * 32 + hh * 16 + 8] = v1;
  };

  v8f acc[4] = {v8f{}, v8f{}, v8f{}, v8f{}};
  stage(0, 0);
  for (int c = 0; c < NCH; ++c) {
    const int buf = c & 1;
    __syncthreads();                     // staging of buf complete
    if (c + 1 < NCH) stage(c + 1, buf ^ 1);  // prefetch next chunk (uniform)
    const v16h af = ld_pair8(&sA[buf][mRowA * 32 + koffA], 16);
    #pragma unroll
    for (int ns = 0; ns < 4; ++ns) {
      const v16h bf = ld_pair8(&sB[buf][(ns * 16 + n) * 32 + koffB], 16);
      acc[ns] = __builtin_amdgcn_wmma_f32_16x16x32_f16(
          false, af, false, bf, (short)0, acc[ns], false, false);
    }
  }

  // epilogue: +bias, store 8 contiguous channels per lane as f16
  const int mrow  = (lane < 16) ? 0 : 8;
  const int mglob = m0blk + wave * 16 + mrow;
  #pragma unroll
  for (int ns = 0; ns < 4; ++ns) {
    v8h o;
    #pragma unroll
    for (int r = 0; r < 8; ++r)
      o[r] = (_Float16)(acc[ns][r] + bias[mglob + r]);
    const long pos = pbase + ns * 16 + n;
    *(v8h*)(out + pos * Cout + mglob) = o;
  }
}

// ---------------------------------------------------------------------------
// 5) BatchNorm (training stats) fold -> scale/shift, then apply+ReLU in place
// ---------------------------------------------------------------------------
__global__ void bn_stats_kernel(const _Float16* __restrict__ act,
                                const float* __restrict__ g,
                                const float* __restrict__ be,
                                float* __restrict__ scale,
                                float* __restrict__ shift,
                                int C, long cnt) {
  __shared__ float sSum[256], sSS[256];
  const int c = blockIdx.x;
  float sum = 0.f, ss = 0.f;
  for (long i = threadIdx.x; i < cnt; i += 256) {
    const float v = (float)act[i * C + c];
    sum += v; ss += v * v;
  }
  sSum[threadIdx.x] = sum; sSS[threadIdx.x] = ss;
  __syncthreads();
  for (int s = 128; s > 0; s >>= 1) {
    if (threadIdx.x < s) {
      sSum[threadIdx.x] += sSum[threadIdx.x + s];
      sSS[threadIdx.x]  += sSS[threadIdx.x + s];
    }
    __syncthreads();
  }
  if (threadIdx.x == 0) {
    const float m  = sSum[0] / (float)cnt;
    const float vr = sSS[0] / (float)cnt - m * m;
    const float sc = g[c] * rsqrtf(vr + 1e-5f);
    scale[c] = sc;
    shift[c] = be[c] - m * sc;
  }
}

__global__ void bn_apply_relu_kernel(_Float16* __restrict__ act,
                                     const float* __restrict__ scale,
                                     const float* __restrict__ shift,
                                     int C, long total) {
  const long i = (long)blockIdx.x * 256 + threadIdx.x;
  if (i >= total) return;
  const int c = (int)(i % C);
  float v = (float)act[i] * scale[c] + shift[c];
  act[i] = (_Float16)(v > 0.f ? v : 0.f);
}

// ---------------------------------------------------------------------------
// 6) 2x2 max pool, NHWC f16
// ---------------------------------------------------------------------------
__global__ void pool_kernel(const _Float16* __restrict__ in,
                            _Float16* __restrict__ out,
                            int H, int W, int C, long total) {
  const long i = (long)blockIdx.x * 256 + threadIdx.x;
  if (i >= total) return;
  const int c = (int)(i % C);
  long p = i / C;
  const int Wo = W >> 1, Ho = H >> 1;
  const int x = (int)(p % Wo); p /= Wo;
  const int y = (int)(p % Ho);
  const int b = (int)(p / Ho);
  const long base = ((long)(b * H + 2 * y) * W + 2 * x) * C + c;
  const float a0 = (float)in[base];
  const float a1 = (float)in[base + C];
  const float a2 = (float)in[base + (long)W * C];
  const float a3 = (float)in[base + (long)W * C + C];
  out[i] = (_Float16)fmaxf(fmaxf(a0, a1), fmaxf(a2, a3));
}

// ---------------------------------------------------------------------------
// FC weight repacks (f32 -> f16, M padded to 512; fc1 re-gathered so its
// K-order matches our NHWC activation flattening)
// ---------------------------------------------------------------------------
__global__ void pack_fc1_kernel(const float* __restrict__ w, _Float16* __restrict__ A1) {
  const long K = 30784;
  const long idx = (long)blockIdx.x * 256 + threadIdx.x;
  if (idx >= 512 * K) return;
  const int m = (int)(idx / K), k = (int)(idx % K);
  float v = 0.f;
  if (m < 500) {
    if (k < 64) {
      v = w[(long)m * K + k];
    } else {
      const int i = k - 64;
      const int y = i >> 11;           // / 2048
      const int x = (i >> 8) & 7;
      const int c = i & 255;
      const int kref = 64 + c * 120 + y * 8 + x;   // reference NCHW flatten
      v = w[(long)m * K + kref];
    }
  }
  A1[idx] = (_Float16)v;
}

__global__ void pack_fc2_kernel(const float* __restrict__ w, _Float16* __restrict__ A2) {
  const long idx = (long)blockIdx.x * 256 + threadIdx.x;
  if (idx >= 512 * 576) return;
  const int m = (int)(idx / 576), k = (int)(idx % 576);
  float v = 0.f;
  if (m < 500 && k < 564) v = w[(long)m * 564 + k];
  A2[idx] = (_Float16)v;
}

// concat(qvec, act3-NHWC-flat) -> xcat1 [64][30784] f16
__global__ void xcat1_kernel(const float* __restrict__ q,
                             const _Float16* __restrict__ act3,
                             _Float16* __restrict__ xc) {
  const long idx = (long)blockIdx.x * 256 + threadIdx.x;
  if (idx >= 64L * 30784) return;
  const int b = (int)(idx / 30784), k = (int)(idx % 30784);
  xc[idx] = (k < 64) ? (_Float16)q[b * 64 + k] : act3[(long)b * 30720 + (k - 64)];
}

// concat(qvec, z[500]) + zero pad -> xcat2 [64][576] f16
__global__ void xcat2_kernel(const float* __restrict__ q,
                             const _Float16* __restrict__ z,
                             _Float16* __restrict__ xc) {
  const long idx = (long)blockIdx.x * 256 + threadIdx.x;
  if (idx >= 64L * 576) return;
  const int b = (int)(idx / 576), k = (int)(idx % 576);
  _Float16 v = (_Float16)0.f;
  if (k < 64) v = (_Float16)q[b * 64 + k];
  else if (k < 564) v = z[(long)b * 512 + (k - 64)];
  xc[idx] = v;
}

// ---------------------------------------------------------------------------
// 7) FC GEMM via WMMA: Z[n][m] = relu(sum_k A[m][k]*X[n][k] + bias[m])
//    grid = (1, 8); block 128; N = 64 batches; double-buffered LDS.
// ---------------------------------------------------------------------------
template <int Kp>
__global__ void __launch_bounds__(128) fc_wmma_kernel(
    const _Float16* __restrict__ A,   // [512][Kp]
    const _Float16* __restrict__ X,   // [64][Kp]
    const float* __restrict__ bias,   // [Mreal]
    _Float16* __restrict__ Z,         // [64][512]
    int Mreal) {
  constexpr int NCH = Kp / 32;
  __shared__ __align__(16) _Float16 sA[2][64 * 32];
  __shared__ __align__(16) _Float16 sB[2][64 * 32];
  const int tid  = threadIdx.x;
  const int lane = tid & 31, wave = tid >> 5;
  const int m0blk = blockIdx.y * 64;
  const int rowS = tid >> 1, hh = tid & 1;
  const int n     = lane & 15;
  const int koffA = (lane < 16) ? 0 : 8;
  const int koffB = (lane < 16) ? 0 : 8;
  const int mRowA = wave * 16 + n;

  auto stage = [&](int c, int buf) {
    const _Float16* asrc = A + (long)(m0blk + rowS) * Kp + c * 32 + hh * 16;
    *(v8h*)&sA[buf][rowS * 32 + hh * 16]     = *(const v8h*)asrc;
    *(v8h*)&sA[buf][rowS * 32 + hh * 16 + 8] = *(const v8h*)(asrc + 8);
    const _Float16* bsrc = X + (long)rowS * Kp + c * 32 + hh * 16;
    *(v8h*)&sB[buf][rowS * 32 + hh * 16]     = *(const v8h*)bsrc;
    *(v8h*)&sB[buf][rowS * 32 + hh * 16 + 8] = *(const v8h*)(bsrc + 8);
  };

  v8f acc[4] = {v8f{}, v8f{}, v8f{}, v8f{}};
  stage(0, 0);
  for (int c = 0; c < NCH; ++c) {
    const int buf = c & 1;
    __syncthreads();
    if (c + 1 < NCH) stage(c + 1, buf ^ 1);
    const v16h af = ld_pair8(&sA[buf][mRowA * 32 + koffA], 16);
    #pragma unroll
    for (int ns = 0; ns < 4; ++ns) {
      const v16h bf = ld_pair8(&sB[buf][(ns * 16 + n) * 32 + koffB], 16);
      acc[ns] = __builtin_amdgcn_wmma_f32_16x16x32_f16(
          false, af, false, bf, (short)0, acc[ns], false, false);
    }
  }

  const int mrow  = (lane < 16) ? 0 : 8;
  const int mglob = m0blk + wave * 16 + mrow;
  #pragma unroll
  for (int ns = 0; ns < 4; ++ns) {
    v8h o;
    #pragma unroll
    for (int r = 0; r < 8; ++r) {
      const float bv = (mglob + r < Mreal) ? bias[mglob + r] : 0.f;
      float v = acc[ns][r] + bv;
      v = v > 0.f ? v : 0.f;             // ReLU (fc1 & fc2 both use it)
      o[r] = (_Float16)v;
    }
    const int batch = ns * 16 + n;
    *(v8h*)(Z + (long)batch * 512 + mglob) = o;
  }
}

// ---------------------------------------------------------------------------
// 8) fc3 (M=2): out[b][j] = concat(q, z2) . fc3_w[j] + fc3_b[j]  (f32)
// ---------------------------------------------------------------------------
__global__ void fc3_kernel(const float* __restrict__ q,
                           const _Float16* __restrict__ z2,
                           const float* __restrict__ w,   // [2][564]
                           const float* __restrict__ b,
                           float* __restrict__ out) {
  const int t = threadIdx.x;
  if (t >= 128) return;
  const int j = t & 1, bb = t >> 1;
  const float* wr = w + j * 564;
  float acc = b[j];
  for (int k = 0; k < 64; ++k)  acc += q[bb * 64 + k] * wr[k];
  for (int i = 0; i < 500; ++i) acc += (float)z2[(long)bb * 512 + i] * wr[64 + i];
  out[bb * 2 + j] = acc;
}

// ---------------------------------------------------------------------------
extern "C" void kernel_launch(void* const* d_in, const int* in_sizes, int n_in,
                              void* d_out, int out_size, void* d_ws, size_t ws_size,
                              hipStream_t stream) {
  (void)in_sizes; (void)n_in; (void)out_size; (void)ws_size;
  const float* x     = (const float*)d_in[0];
  const float* qv    = (const float*)d_in[1];
  const float* basis = (const float*)d_in[2];
  const float* w1a   = (const float*)d_in[3];
  const float* b1a   = (const float*)d_in[4];
  const float* g1a   = (const float*)d_in[5];
  const float* be1a  = (const float*)d_in[6];
  const float* w1b   = (const float*)d_in[7];
  const float* b1b   = (const float*)d_in[8];
  const float* g1b   = (const float*)d_in[9];
  const float* be1b  = (const float*)d_in[10];
  const float* w2a   = (const float*)d_in[11];
  const float* b2a   = (const float*)d_in[12];
  const float* g2a   = (const float*)d_in[13];
  const float* be2a  = (const float*)d_in[14];
  const float* w3a   = (const float*)d_in[15];
  const float* b3a   = (const float*)d_in[16];
  const float* g3a   = (const float*)d_in[17];
  const float* be3a  = (const float*)d_in[18];
  const float* fc1w  = (const float*)d_in[19];
  const float* fc1b  = (const float*)d_in[20];
  const float* fc2w  = (const float*)d_in[21];
  const float* fc2b  = (const float*)d_in[22];
  const float* fc3w  = (const float*)d_in[23];
  const float* fc3b  = (const float*)d_in[24];
  float* out = (float*)d_out;

  char* ws = (char*)d_ws;
  size_t off = 0;
  auto alc = [&](size_t bytes) -> void* {
    void* p = ws + off;
    off += (bytes + 255) & ~(size_t)255;
    return p;
  };
  float*    xd   = (float*)alc(64ull * 64 * 1024 * 4);
  float*    feat = (float*)alc(64ull * 120 * 64 * 4);
  _Float16* bufA = (_Float16*)alc(64ull * 120 * 64 * 64 * 2);
  _Float16* bufB = (_Float16*)alc(64ull * 120 * 64 * 64 * 2);
  _Float16* wq1b = (_Float16*)alc(64ull * 1600 * 2);
  _Float16* wq2a = (_Float16*)alc(128ull * 1600 * 2);
  _Float16* wq3a = (_Float16*)alc(256ull * 3200 * 2);
  _Float16* A1   = (_Float16*)alc(512ull * 30784 * 2);
  _Float16* A2   = (_Float16*)alc(512ull * 576 * 2);
  _Float16* xc1  = (_Float16*)alc(64ull * 30784 * 2);
  _Float16* xc2  = (_Float16*)alc(64ull * 576 * 2);
  _Float16* z1   = (_Float16*)alc(64ull * 512 * 2);
  _Float16* z2   = (_Float16*)alc(64ull * 512 * 2);
  float*    scl  = (float*)alc(256 * 4);
  float*    sft  = (float*)alc(256 * 4);

  // ---- weight prep (deterministic, every call) ----
  pack_w_kernel<<<(64 * 1600 + 255) / 256, 256, 0, stream>>>(w1b, wq1b, 64, 64);
  pack_w_kernel<<<(128 * 1600 + 255) / 256, 256, 0, stream>>>(w2a, wq2a, 128, 64);
  pack_w_kernel<<<(256 * 3200 + 255) / 256, 256, 0, stream>>>(w3a, wq3a, 256, 128);
  pack_fc1_kernel<<<(int)((512L * 30784 + 255) / 256), 256, 0, stream>>>(fc1w, A1);
  pack_fc2_kernel<<<(512 * 576 + 255) / 256, 256, 0, stream>>>(fc2w, A2);

  // ---- DCT + soft histogram ----
  dct_kernel<<<16384, 256, 0, stream>>>(x, basis, xd);
  hist_kernel<<<4096, 128, 0, stream>>>(xd, feat);

  // ---- conv1a (direct) + BN + ReLU ----
  const long n1 = 64L * 120 * 64 * 64;
  conv1a_kernel<<<(int)((n1 + 255) / 256), 256, 0, stream>>>(feat, w1a, b1a, bufA);
  bn_stats_kernel<<<64, 256, 0, stream>>>(bufA, g1a, be1a, scl, sft, 64, 64L * 120 * 64);
  bn_apply_relu_kernel<<<(int)((n1 + 255) / 256), 256, 0, stream>>>(bufA, scl, sft, 64, n1);

  // ---- conv1b (WMMA) + BN + ReLU + pool -> [B,60,32,64] ----
  conv_wmma_kernel<120, 64, 64, 64>
      <<<dim3(7680, 1), 128, 0, stream>>>(bufA, wq1b, b1b, bufB);
  bn_stats_kernel<<<64, 256, 0, stream>>>(bufB, g1b, be1b, scl, sft, 64, 64L * 120 * 64);
  bn_apply_relu_kernel<<<(int)((n1 + 255) / 256), 256, 0, stream>>>(bufB, scl, sft, 64, n1);
  const long p1 = 64L * 60 * 32 * 64;
  pool_kernel<<<(int)((p1 + 255) / 256), 256, 0, stream>>>(bufB, bufA, 120, 64, 64, p1);

  // ---- conv2a (WMMA) + BN + ReLU + pool -> [B,30,16,128] ----
  conv_wmma_kernel<60, 32, 64, 128>
      <<<dim3(1920, 2), 128, 0, stream>>>(bufA, wq2a, b2a, bufB);
  const long n2 = 64L * 60 * 32 * 128;
  bn_stats_kernel<<<128, 256, 0, stream>>>(bufB, g2a, be2a, scl, sft, 128, 64L * 60 * 32);
  bn_apply_relu_kernel<<<(int)((n2 + 255) / 256), 256, 0, stream>>>(bufB, scl, sft, 128, n2);
  const long p2 = 64L * 30 * 16 * 128;
  pool_kernel<<<(int)((p2 + 255) / 256), 256, 0, stream>>>(bufB, bufA, 60, 32, 128, p2);

  // ---- conv3a (WMMA) + BN + ReLU + pool -> [B,15,8,256] ----
  conv_wmma_kernel<30, 16, 128, 256>
      <<<dim3(480, 4), 128, 0, stream>>>(bufA, wq3a, b3a, bufB);
  const long n3 = 64L * 30 * 16 * 256;
  bn_stats_kernel<<<256, 256, 0, stream>>>(bufB, g3a, be3a, scl, sft, 256, 64L * 30 * 16);
  bn_apply_relu_kernel<<<(int)((n3 + 255) / 256), 256, 0, stream>>>(bufB, scl, sft, 256, n3);
  const long p3 = 64L * 15 * 8 * 256;
  pool_kernel<<<(int)((p3 + 255) / 256), 256, 0, stream>>>(bufB, bufA, 30, 16, 256, p3);

  // ---- MLP head ----
  xcat1_kernel<<<(int)((64L * 30784 + 255) / 256), 256, 0, stream>>>(qv, bufA, xc1);
  fc_wmma_kernel<30784><<<dim3(1, 8), 128, 0, stream>>>(A1, xc1, fc1b, z1, 500);
  xcat2_kernel<<<(64 * 576 + 255) / 256, 256, 0, stream>>>(qv, z1, xc2);
  fc_wmma_kernel<576><<<dim3(1, 8), 128, 0, stream>>>(A2, xc2, fc2b, z2, 500);
  fc3_kernel<<<1, 128, 0, stream>>>(qv, z2, fc3w, fc3b, out);
}
